// DistBlock_24747601560227
// MI455X (gfx1250) — matrix operand order repaired and checked by reading, not compiled, not run on hardware
//
#include <hip/hip_runtime.h>

#define Bb   64
#define Ss   197
#define HIDh 768
#define NHh  12
#define DHd  64
#define MSr  (Bb*Ss)      /* 12608 rows, divisible by 16 */
#define SP   224          /* S padded: 14*16 = 7*32 */
#define SPC  (SP/32)      /* 7 K-chunks of 32 */

typedef __attribute__((ext_vector_type(16))) __bf16 bf16x16;
typedef __attribute__((ext_vector_type(8)))  float  f32x8;
typedef unsigned short u16;
typedef unsigned int   u32;

__device__ __forceinline__ u16 f2bf(float f) {
  u32 u; __builtin_memcpy(&u, &f, 4);
  u = (u + 0x7FFFu + ((u >> 16) & 1u)) >> 16;
  return (u16)u;
}
__device__ __forceinline__ f32x8 wmma_bf16(bf16x16 a, bf16x16 b, f32x8 c) {
  return __builtin_amdgcn_wmma_f32_16x16x32_bf16(false, a, false, b, (short)0, c, false, false);
}
/* Build a 16-element bf16 fragment from two 16-byte runs (global or LDS). */
__device__ __forceinline__ bf16x16 ld_frag(const u16* p0, const u16* p1) {
  union { uint4 u[2]; bf16x16 v; } t;
  t.u[0] = *(const uint4*)p0;
  t.u[1] = *(const uint4*)p1;
  return t.v;
}

/* ---------------- f32 -> bf16 conversion ---------------- */
__global__ void cvt_f32_bf16(const float* __restrict__ in, u16* __restrict__ out, size_t n) {
  size_t i = (size_t)blockIdx.x * blockDim.x + threadIdx.x;
  size_t st = (size_t)gridDim.x * blockDim.x;
  for (; i < n; i += st) out[i] = f2bf(in[i]);
}

/* f32 W[K,N] -> bf16 W'[(k/32)*N + n][k%32]  (B-fragment friendly) */
__global__ void cvt_swizzle_w(const float* __restrict__ in, u16* __restrict__ out, int K, int N) {
  size_t tot = (size_t)K * N;
  size_t i = (size_t)blockIdx.x * blockDim.x + threadIdx.x;
  size_t st = (size_t)gridDim.x * blockDim.x;
  for (; i < tot; i += st) {
    int kk = (int)(i / N), n = (int)(i - (size_t)kk * N);
    out[((size_t)(kk >> 5) * N + n) * 32 + (kk & 31)] = f2bf(in[i]);
  }
}

/* ---------------- generic bf16 WMMA GEMM ----------------
 * C[M,N] = A[M,K] @ W + bias  (W pre-swizzled K-chunked)
 * MODE 0: store bf16 padded head-major [bh][SP][DH]; rowsum += y^2
 * MODE 1: z=elu(y)+1; store sqrt(max(z,EPS)) padded head-major; rowsum += z
 * MODE 2: z=elu(y)+1; store z in swizzled-V layout
 * MODE 3: store f32 row-major [M,N]
 * MODE 4: store y in swizzled-V layout
 * Grid (M/16, N/256), block 128 = 4 waves, wave owns a 16x64 strip (one head).
 */
template<int MODE>
__global__ __launch_bounds__(128) void gemm_bf16_kernel(
    const u16* __restrict__ A, const u16* __restrict__ Wsw, const float* __restrict__ bias,
    u16* __restrict__ outBF, float* __restrict__ outF, float* __restrict__ rowsum,
    int M, int K, int N)
{
  const int lane = threadIdx.x & 31;
  const int wave = threadIdx.x >> 5;
  const int hf   = lane >> 4;
  const int l16  = lane & 15;
  const int mbase = blockIdx.x * 16;
  const int nbase = blockIdx.y * 256 + wave * 64;

  f32x8 acc[4] = {};
  const u16* Arow = A + (size_t)(mbase + l16) * K;

  for (int kk = 0; kk < K; kk += 32) {
    const u16* ap = Arow + kk + hf * 16;
    bf16x16 af = ld_frag(ap, ap + 8);
    if (kk + 32 < K) {
      __builtin_prefetch(Arow + kk + 32, 0, 1);
      __builtin_prefetch(Wsw + ((size_t)((kk >> 5) + 1) * N + nbase + l16) * 32, 0, 1);
    }
    #pragma unroll
    for (int t = 0; t < 4; t++) {
      const u16* wp = Wsw + ((size_t)(kk >> 5) * N + (nbase + t * 16 + l16)) * 32 + hf * 16;
      acc[t] = wmma_bf16(af, ld_frag(wp, wp + 8), acc[t]);
    }
  }

  float rs[8];
  #pragma unroll
  for (int v = 0; v < 8; v++) rs[v] = 0.f;

  #pragma unroll
  for (int t = 0; t < 4; t++) {
    int n = nbase + t * 16 + l16;
    float bv = bias[n];
    #pragma unroll
    for (int v = 0; v < 8; v++) {
      int m = mbase + v + hf * 8;
      float y = acc[t][v] + bv;
      if (MODE == 3) {
        outF[(size_t)m * N + n] = y;
      } else {
        int b = m / Ss, s = m - b * Ss;
        int h = n >> 6, d = n & 63;
        size_t bh = (size_t)(b * NHh + h);
        float st;
        if (MODE == 0) { st = y; rs[v] += y * y; }
        else if (MODE == 4) { st = y; }
        else {
          float z = (y > 0.f) ? (y + 1.f) : __expf(y);
          if (MODE == 1) { rs[v] += z; st = sqrtf(fmaxf(z, 1e-24f)); }
          else st = z;
        }
        size_t o;
        if (MODE == 2 || MODE == 4)
          o = ((bh * SPC + (s >> 5)) * DHd + d) * 32 + (s & 31);     // swizzled V
        else
          o = (bh * SP + s) * DHd + d;                                // padded head-major
        outBF[o] = f2bf(st);
      }
    }
  }

  if (MODE == 0 || MODE == 1) {
    int h = nbase >> 6;
    #pragma unroll
    for (int v = 0; v < 8; v++) {
      float r = rs[v];
      r += __shfl_xor(r, 1, 16);
      r += __shfl_xor(r, 2, 16);
      r += __shfl_xor(r, 4, 16);
      r += __shfl_xor(r, 8, 16);
      if (l16 == 0) {
        int m = mbase + v + hf * 8;
        int b = m / Ss, s = m - b * Ss;
        atomicAdd(&rowsum[(size_t)(b * NHh + h) * SP + s], r);
      }
    }
  }
}

/* ---------------- attention: scores, softmax, P@V ---------------- */
__global__ __launch_bounds__(32) void attn_kernel(
    const u16* __restrict__ mq, const u16* __restrict__ mk,
    const u16* __restrict__ sq, const u16* __restrict__ sk,
    const u16* __restrict__ mvsw, const u16* __restrict__ cvsw,
    const float* __restrict__ rowq, const float* __restrict__ rowk,
    u16* __restrict__ ctxm, u16* __restrict__ ctxc)
{
  __shared__ float sc[16][SP];
  __shared__ u16   pr[16][SP];
  __shared__ u16   pr2[16][SP];

  const int lane = threadIdx.x;
  const int hf = lane >> 4, l16 = lane & 15;
  const int qt = blockIdx.x;            // 0..12
  const int bh = blockIdx.y;            // 0..B*NH-1
  const int b  = bh / NHh, h = bh - b * NHh;
  const size_t baseR = (size_t)bh * SP * DHd;   // padded row-major per (b,h)
  const float* rqp = rowq + (size_t)bh * SP;
  const float* rkp = rowk + (size_t)bh * SP;

  // Q-side A fragments (mean and sqrt-cov); pad rows are zero-filled
  bf16x16 aqm[2], aqs[2];
  {
    const u16* mqr = mq + baseR + (size_t)(qt * 16 + l16) * DHd;
    const u16* sqr = sq + baseR + (size_t)(qt * 16 + l16) * DHd;
    #pragma unroll
    for (int dc = 0; dc < 2; dc++) {
      int d0 = dc * 32 + hf * 16;
      aqm[dc] = ld_frag(mqr + d0, mqr + d0 + 8);
      aqs[dc] = ld_frag(sqr + d0, sqr + d0 + 8);
    }
  }

  // scores = (2*(m.mT + s.sT) - rowq - rowk) / 8
  for (int kt = 0; kt < SP / 16; kt++) {
    const u16* mkr = mk + baseR + (size_t)(kt * 16 + l16) * DHd;
    const u16* skr = sk + baseR + (size_t)(kt * 16 + l16) * DHd;
    f32x8 acc = {};
    #pragma unroll
    for (int dc = 0; dc < 2; dc++) {
      int d0 = dc * 32 + hf * 16;
      acc = wmma_bf16(aqm[dc], ld_frag(mkr + d0, mkr + d0 + 8), acc);
      acc = wmma_bf16(aqs[dc], ld_frag(skr + d0, skr + d0 + 8), acc);
    }
    int kc = kt * 16 + l16;
    float rk = rkp[kc];
    #pragma unroll
    for (int v = 0; v < 8; v++) {
      int ql = v + 8 * hf;
      float rq = rqp[qt * 16 + ql];
      float val = (2.f * acc[v] - rq - rk) * 0.125f;
      sc[ql][kc] = (kc < Ss) ? val : -3.0e38f;
    }
  }
  __syncthreads();

  // per-row softmax (lanes 0..15); build P and P^2 in bf16, zero the padding
  if (lane < 16) {
    float mx = -3.0e38f;
    for (int c = 0; c < Ss; c++) mx = fmaxf(mx, sc[lane][c]);
    float sum = 0.f;
    for (int c = 0; c < Ss; c++) { float e = __expf(sc[lane][c] - mx); sc[lane][c] = e; sum += e; }
    float inv = 1.f / sum;
    for (int c = 0; c < Ss; c++) {
      float p = sc[lane][c] * inv;
      pr[lane][c]  = f2bf(p);
      pr2[lane][c] = f2bf(p * p);
    }
    for (int c = Ss; c < SP; c++) { pr[lane][c] = 0; pr2[lane][c] = 0; }
  }
  __syncthreads();

  // ctx_mean = P @ V_mean ; ctx_cov = (P*P) @ V_cov   (V pre-swizzled, pads zero)
  #pragma unroll
  for (int nt = 0; nt < 4; nt++) {
    f32x8 am = {}, ac = {};
    for (int kc = 0; kc < SPC; kc++) {
      int k0 = kc * 32 + hf * 16;
      bf16x16 ap  = ld_frag(&pr[l16][k0],  &pr[l16][k0 + 8]);
      bf16x16 ap2 = ld_frag(&pr2[l16][k0], &pr2[l16][k0 + 8]);
      const u16* vp = mvsw + (((size_t)bh * SPC + kc) * DHd + nt * 16 + l16) * 32 + hf * 16;
      const u16* cp = cvsw + (((size_t)bh * SPC + kc) * DHd + nt * 16 + l16) * 32 + hf * 16;
      am = wmma_bf16(ap,  ld_frag(vp, vp + 8), am);
      ac = wmma_bf16(ap2, ld_frag(cp, cp + 8), ac);
    }
    #pragma unroll
    for (int v = 0; v < 8; v++) {
      int qg = qt * 16 + v + 8 * hf;
      if (qg < Ss) {
        size_t o = ((size_t)b * Ss + qg) * HIDh + h * DHd + nt * 16 + l16;
        ctxm[o] = f2bf(am[v]);
        ctxc[o] = f2bf(ac[v]);
      }
    }
  }
}

/* ---------------- host orchestration ---------------- */
extern "C" void kernel_launch(void* const* d_in, const int* in_sizes, int n_in,
                              void* d_out, int out_size, void* d_ws, size_t ws_size,
                              hipStream_t stream) {
  (void)in_sizes; (void)n_in; (void)out_size; (void)ws_size;

  const float* xm = (const float*)d_in[0];
  const float* xc = (const float*)d_in[1];
  const float* Wf[8]; const float* Bf[8];
  for (int i = 0; i < 8; i++) { Wf[i] = (const float*)d_in[2 + 2 * i]; Bf[i] = (const float*)d_in[3 + 2 * i]; }

  const size_t WN  = (size_t)HIDh * HIDh;          // weight elems
  const size_t XN  = (size_t)MSr * HIDh;           // activation elems
  const size_t HNp = (size_t)Bb * NHh * SP * DHd;  // padded head-major / swizzled-V elems
  const size_t RNp = (size_t)Bb * NHh * SP;        // padded rowsum elems

  char* p = (char*)d_ws;
  auto carve = [&](size_t bytes) -> void* {
    void* r = (void*)p; p += (bytes + 255) & ~(size_t)255; return r;
  };
  u16* wbf[8]; for (int i = 0; i < 8; i++) wbf[i] = (u16*)carve(WN * 2);
  u16* xmb  = (u16*)carve(XN * 2);
  u16* xcb  = (u16*)carve(XN * 2);
  u16* mq   = (u16*)carve(HNp * 2);
  u16* mk   = (u16*)carve(HNp * 2);
  u16* sq   = (u16*)carve(HNp * 2);
  u16* sk   = (u16*)carve(HNp * 2);
  u16* mvsw = (u16*)carve(HNp * 2);
  u16* cvsw = (u16*)carve(HNp * 2);
  float* rowq = (float*)carve(RNp * 4);
  float* rowk = (float*)carve(RNp * 4);
  u16* ctxm = (u16*)carve(XN * 2);
  u16* ctxc = (u16*)carve(XN * 2);

  // 1) convert + swizzle weights, convert inputs
  for (int i = 0; i < 8; i++)
    cvt_swizzle_w<<<1024, 256, 0, stream>>>(Wf[i], wbf[i], HIDh, HIDh);
  cvt_f32_bf16<<<4096, 256, 0, stream>>>(xm, xmb, XN);
  cvt_f32_bf16<<<4096, 256, 0, stream>>>(xc, xcb, XN);

  // 2) zero-fill padded operand arrays + rowsums (pad rows must be 0)
  hipMemsetAsync(mq,   0, HNp * 2, stream);
  hipMemsetAsync(mk,   0, HNp * 2, stream);
  hipMemsetAsync(sq,   0, HNp * 2, stream);
  hipMemsetAsync(sk,   0, HNp * 2, stream);
  hipMemsetAsync(mvsw, 0, HNp * 2, stream);
  hipMemsetAsync(cvsw, 0, HNp * 2, stream);
  hipMemsetAsync(rowq, 0, RNp * 4, stream);
  hipMemsetAsync(rowk, 0, RNp * 4, stream);

  // 3) projections (fused rowsum / elu / sqrt / swizzled-V epilogues)
  dim3 gg(MSr / 16, HIDh / 256);   // (788, 3)
  gemm_bf16_kernel<0><<<gg, 128, 0, stream>>>(xmb, wbf[0], Bf[0], mq,   nullptr, rowq, MSr, HIDh, HIDh);
  gemm_bf16_kernel<1><<<gg, 128, 0, stream>>>(xcb, wbf[1], Bf[1], sq,   nullptr, rowq, MSr, HIDh, HIDh);
  gemm_bf16_kernel<0><<<gg, 128, 0, stream>>>(xmb, wbf[2], Bf[2], mk,   nullptr, rowk, MSr, HIDh, HIDh);
  gemm_bf16_kernel<1><<<gg, 128, 0, stream>>>(xcb, wbf[3], Bf[3], sk,   nullptr, rowk, MSr, HIDh, HIDh);
  gemm_bf16_kernel<4><<<gg, 128, 0, stream>>>(xmb, wbf[4], Bf[4], mvsw, nullptr, nullptr, MSr, HIDh, HIDh);
  gemm_bf16_kernel<2><<<gg, 128, 0, stream>>>(xcb, wbf[5], Bf[5], cvsw, nullptr, nullptr, MSr, HIDh, HIDh);

  // 4) attention (one wave per (b,h,q-tile))
  dim3 ga((Ss + 15) / 16, Bb * NHh);   // (13, 768)
  attn_kernel<<<ga, 32, 0, stream>>>(mq, mk, sq, sk, mvsw, cvsw, rowq, rowk, ctxm, ctxc);

  // 5) output projections -> f32 into d_out (mean first, then cov)
  float* out_mean = (float*)d_out;
  float* out_cov  = out_mean + XN;
  gemm_bf16_kernel<3><<<gg, 128, 0, stream>>>(ctxm, wbf[6], Bf[6], nullptr, out_mean, nullptr, MSr, HIDh, HIDh);
  gemm_bf16_kernel<3><<<gg, 128, 0, stream>>>(ctxc, wbf[7], Bf[7], nullptr, out_cov,  nullptr, MSr, HIDh, HIDh);
}
